// Model_88991722373632
// MI455X (gfx1250) — compile-verified, matrix operand order checked by reading, not run on hardware
//
#include <hip/hip_runtime.h>
#include <stdint.h>

#define BB   64
#define TT   200
#define INW  100002
#define HH   128
#define G4   512     // 4*H

#if defined(__has_builtin)
#  if __has_builtin(__builtin_amdgcn_global_load_async_to_lds_b128) && \
      __has_builtin(__builtin_amdgcn_s_wait_asynccnt)
#    define USE_ASYNC 1
#  else
#    define USE_ASYNC 0
#  endif
#else
#  define USE_ASYNC 0
#endif

typedef _Float16 v8h  __attribute__((ext_vector_type(8)));
typedef _Float16 v16h __attribute__((ext_vector_type(16)));
typedef float    v4f  __attribute__((ext_vector_type(4)));
typedef float    v8f  __attribute__((ext_vector_type(8)));

#if USE_ASYNC
// builtin signature (from compiler diagnostics):
//   (AS(1) int4* gsrc, AS(3) int4* ldst, imm i32 offset, imm i32 cpol)
typedef int i32x4_t __attribute__((vector_size(16)));
typedef __attribute__((address_space(1))) i32x4_t gvec_t;
typedef __attribute__((address_space(3))) i32x4_t lvec_t;
#endif

__device__ __forceinline__ float sigm(float x) {
    x = fminf(fmaxf(x, -30.f), 30.f);
    return 1.0f / (1.0f + __expf(-x));
}
__device__ __forceinline__ float tanhx(float x) {
    x = fminf(fmaxf(x, -15.f), 15.f);
    float e = __expf(2.0f * x);
    return (e - 1.0f) / (e + 1.0f);
}

// Async-DMA one step's 32 floats (128 B) for this lane into its private LDS
// region. IOFFSET (literal immediate) is added to BOTH global and LDS
// addresses (ISA 08 §4.4).
__device__ __forceinline__ void gi_prefetch8(const float* gsrc, float* ldst) {
#if USE_ASYNC
    gvec_t* gp = (gvec_t*)(uintptr_t)gsrc;
    lvec_t* lp = (lvec_t*)(uint32_t)(uintptr_t)ldst;
    __builtin_amdgcn_global_load_async_to_lds_b128(gp, lp, 0,   0);
    __builtin_amdgcn_global_load_async_to_lds_b128(gp, lp, 16,  0);
    __builtin_amdgcn_global_load_async_to_lds_b128(gp, lp, 32,  0);
    __builtin_amdgcn_global_load_async_to_lds_b128(gp, lp, 48,  0);
    __builtin_amdgcn_global_load_async_to_lds_b128(gp, lp, 64,  0);
    __builtin_amdgcn_global_load_async_to_lds_b128(gp, lp, 80,  0);
    __builtin_amdgcn_global_load_async_to_lds_b128(gp, lp, 96,  0);
    __builtin_amdgcn_global_load_async_to_lds_b128(gp, lp, 112, 0);
#endif
}

// ---------------------------------------------------------------- lengths
__global__ void lens_kernel(const int* __restrict__ ids, int* __restrict__ lens) {
    int b = threadIdx.x;
    if (b < BB) {
        int c = 0;
        for (int t = 0; t < TT; ++t) c += (ids[b * TT + t] != 0);
        lens[b] = c;
    }
}

// ------------------------------------------------- embedding gather (both dirs)
// Writes gi in the scan kernel's lane-permuted layout:
//   gi[(dir*TT + t)*32768 + (bg*256 + tid)*32 + (gs*8 + r)]
//     = w_ih[g, id(b)] + b_ih[g] + b_hh[g]
// where tid = w*32 + lhi*16 + l15, b = bg*16 + 8*lhi + r, g = gs*128 + 16*w + l15.
__global__ __launch_bounds__(256) void gather_kernel(
        const int* __restrict__ ids, const int* __restrict__ lens,
        const float* __restrict__ w_ih_f, const float* __restrict__ b_ih_f,
        const float* __restrict__ b_hh_f,
        const float* __restrict__ w_ih_b, const float* __restrict__ b_ih_b,
        const float* __restrict__ b_hh_b,
        float* __restrict__ gi) {
    int idx = blockIdx.x;             // (dir*TT + t)*4 + bg
    int bg  = idx & 3;
    int dt  = idx >> 2;
    int t   = dt % TT;
    int dir = dt / TT;

    int tid  = threadIdx.x;
    int lane = tid & 31;
    int w    = tid >> 5;
    int l15  = lane & 15;
    int lhi  = lane >> 4;

    const float* W  = dir ? w_ih_b : w_ih_f;
    const float* bi = dir ? b_ih_b : b_ih_f;
    const float* bh = dir ? b_hh_b : b_hh_f;

    int idv[8];
#pragma unroll
    for (int r = 0; r < 8; ++r) {
        int b = bg * 16 + 8 * lhi + r;
        int id;
        if (dir == 0) {
            id = ids[b * TT + t];
        } else {
            int len = lens[b];
            int rt = (t < len) ? (len - 1 - t) : t;   // packed-seq reversal
            id = ids[b * TT + rt];
        }
        idv[r] = id;
    }

    float* o = gi + (size_t)dt * (BB * G4) + (size_t)(bg * 256 + tid) * 32;
#pragma unroll
    for (int gs = 0; gs < 4; ++gs) {
        int g = gs * 128 + 16 * w + l15;
        float add = bi[g] + bh[g];
        const float* col = W + (size_t)g * INW;
#pragma unroll
        for (int r = 0; r < 8; ++r)
            o[gs * 8 + r] = col[idv[r]] + add;
    }
}

// ------------------------------------------------------------- LSTM scan
// 8 blocks = 2 dirs x 4 batch-groups of 16 rows; 256 threads = 8 waves.
// Wave w owns N-tiles {w, w+8, w+16, w+24} = gates i,f,g,o for hidden block w,
// so the LSTM cell runs directly on WMMA accumulators (no gates LDS).
// h ping-pong in static LDS; gi DMA'd 2 steps ahead into dynamic LDS via
// global_load_async_to_lds_b128 (ASYNCcnt); manual 2x unroll (no buffer movs).
__global__ __launch_bounds__(256) void scan_kernel(const float* __restrict__ gi,
                                                   const float* __restrict__ w_hh_f,
                                                   const float* __restrict__ w_hh_b,
                                                   const int* __restrict__ lens,
                                                   float* __restrict__ rep) {
    __shared__ _Float16 hb[2 * 16 * HH];      // 8 KB static: h ping-pong
    extern __shared__ float gib[];            // 64 KB dynamic: gi staging x2
    const int HSZ = 16 * HH;
    const int GSZ = 256 * 32;

    const int blk = blockIdx.x;
    const int dir = blk >> 2;
    const int bg  = blk & 3;

    const int tid  = threadIdx.x;
    const int lane = tid & 31;
    const int w    = tid >> 5;       // 0..7
    const int l15  = lane & 15;
    const int lhi  = lane >> 4;

    // ---- preload B fragments: 4 gate-tiles x 4 k-steps of w_hh.T (f16) ----
    const float* w_hh = dir ? w_hh_b : w_hh_f;
    v16h bfrag[4][4];
#pragma unroll
    for (int gs = 0; gs < 4; ++gs) {
        int N = 16 * (gs * 8 + w) + l15;     // = gs*128 + 16*w + l15
        const float* rp = w_hh + (size_t)N * HH;
#pragma unroll
        for (int k = 0; k < 4; ++k) {
            int c0 = 32 * k + 8 * lhi;
            v4f f0 = *(const v4f*)(rp + c0);
            v4f f1 = *(const v4f*)(rp + c0 + 4);
            v4f f2 = *(const v4f*)(rp + c0 + 16);
            v4f f3 = *(const v4f*)(rp + c0 + 20);
            v16h bb;
#pragma unroll
            for (int i = 0; i < 4; ++i) {
                bb[i]      = (_Float16)f0[i];
                bb[4 + i]  = (_Float16)f1[i];
                bb[8 + i]  = (_Float16)f2[i];
                bb[12 + i] = (_Float16)f3[i];
            }
            bfrag[gs][k] = bb;
        }
    }

    // ---- per-lane state: 8 batch rows (M = 8*lhi + r), hidden j = 16*w + l15 ----
    float cst[8], rmax[8];
    int   ln[8];
#pragma unroll
    for (int r = 0; r < 8; ++r) {
        int b = bg * 16 + 8 * lhi + r;
        ln[r]   = lens[b];
        cst[r]  = 0.0f;
        rmax[r] = (ln[r] < TT) ? 0.0f : -3.4e38f;   // masked zeros iff len < T
    }

    for (int i = tid; i < HSZ; i += 256) hb[i] = (_Float16)0.0f;

    const float* gbase = gi + (size_t)dir * TT * (BB * G4) + (size_t)(bg * 256 + tid) * 32;
    float* greg0 = gib + tid * 32;
    float* greg1 = gib + GSZ + tid * 32;

#if USE_ASYNC
    // prologue: 2-deep pipeline (16 async b128 in flight)
    gi_prefetch8(gbase, greg0);
    gi_prefetch8(gbase + (size_t)(BB * G4), greg1);
#endif
    __syncthreads();

    auto step = [&](int t, const _Float16* hin, _Float16* hout, float* greg) {
        // ---- A fragments from current h (16-bit A layout) ----
        v16h a[4];
#pragma unroll
        for (int k = 0; k < 4; ++k) {
            const _Float16* hp = &hin[l15 * HH + 32 * k + 8 * lhi];
            v8h lo = *(const v8h*)hp;
            v8h hi = *(const v8h*)(hp + 16);
            v16h aa;
#pragma unroll
            for (int i = 0; i < 8; ++i) { aa[i] = lo[i]; aa[8 + i] = hi[i]; }
            a[k] = aa;
        }

        // ---- gi for this step: one v8f (32 B) per gate block ----
        v8f gvv8[4];
#if USE_ASYNC
        __builtin_amdgcn_s_wait_asynccnt(8);   // retire this step's 8 DMAs
#pragma unroll
        for (int gs = 0; gs < 4; ++gs) gvv8[gs] = *(const v8f*)(greg + gs * 8);
#else
        const float* gsrc = gbase + (size_t)t * (BB * G4);
#pragma unroll
        for (int gs = 0; gs < 4; ++gs) gvv8[gs] = *(const v8f*)(gsrc + gs * 8);
#endif

        // ---- gates = h @ w_hh.T : 16 WMMAs, all four gates in registers ----
        v8f acc[4];
#pragma unroll
        for (int gs = 0; gs < 4; ++gs) acc[gs] = {};
#pragma unroll
        for (int k = 0; k < 4; ++k)
#pragma unroll
            for (int gs = 0; gs < 4; ++gs)
                acc[gs] = __builtin_amdgcn_wmma_f32_16x16x32_f16(
                    false, a[k], false, bfrag[gs][k], (short)0, acc[gs], false, false);

        // ---- gate preactivations: whole-vector adds (register-aligned pk_add) ----
        v8f sI = acc[0] + gvv8[0];
        v8f sF = acc[1] + gvv8[1];
        v8f sG = acc[2] + gvv8[2];
        v8f sO = acc[3] + gvv8[3];

        // ---- LSTM cell: D layout M = r + 8*lhi, N = l15 ----
#pragma unroll
        for (int r = 0; r < 8; ++r) {
            float cc = sigm(sF[r]) * cst[r] + sigm(sI[r]) * tanhx(sG[r]);
            cst[r] = cc;
            float h = sigm(sO[r]) * tanhx(cc);
            if (t < ln[r]) rmax[r] = fmaxf(rmax[r], h);
            hout[(8 * lhi + r) * HH + 16 * w + l15] = (_Float16)h;
        }

        __syncthreads();   // waits dscnt: h visible, gi-LDS reads retired

#if USE_ASYNC
        // refill the buffer just consumed with step t+2 (clamped: invariant
        // 16 in flight so s_wait_asynccnt(8) is always the right threshold)
        int tn = t + 2;
        if (tn >= TT) tn = TT - 1;
        gi_prefetch8(gbase + (size_t)tn * (BB * G4), greg);
#endif
    };

    for (int t = 0; t < TT; t += 2) {
        step(t,     hb,       hb + HSZ, greg0);
        step(t + 1, hb + HSZ, hb,       greg1);
    }

    // ---- rep[b][dir*128 + j] = running max ----
#pragma unroll
    for (int r = 0; r < 8; ++r)
        rep[(size_t)(bg * 16 + 8 * lhi + r) * 256 + dir * HH + 16 * w + l15] = rmax[r];
}

// ---------------------------------------------------------------- final FC
__global__ void fc_kernel(const float* __restrict__ rep, const float* __restrict__ fc_w,
                          const float* __restrict__ fc_b, float* __restrict__ out) {
    int tid = threadIdx.x;
    if (tid < 128) {
        int b = tid >> 1, o = tid & 1;
        float s = fc_b[o];
        for (int j = 0; j < 256; ++j) s += rep[b * 256 + j] * fc_w[o * 256 + j];
        out[b * 2 + o] = s;
    }
}

extern "C" void kernel_launch(void* const* d_in, const int* in_sizes, int n_in,
                              void* d_out, int out_size, void* d_ws, size_t ws_size,
                              hipStream_t stream) {
    const int*   ids    = (const int*)d_in[0];
    const float* w_ih_f = (const float*)d_in[1];
    const float* w_hh_f = (const float*)d_in[2];
    const float* b_ih_f = (const float*)d_in[3];
    const float* b_hh_f = (const float*)d_in[4];
    const float* w_ih_b = (const float*)d_in[5];
    const float* w_hh_b = (const float*)d_in[6];
    const float* b_ih_b = (const float*)d_in[7];
    const float* b_hh_b = (const float*)d_in[8];
    const float* fc_w   = (const float*)d_in[9];
    const float* fc_b   = (const float*)d_in[10];
    float* out = (float*)d_out;

    char*  ws   = (char*)d_ws;
    int*   lens = (int*)ws;                                  // 64 ints
    float* rep  = (float*)(ws + 256);                        // 64*256 f32 (64 KB)
    float* gi   = (float*)(ws + 256 + 64 * 256 * 4);         // 2*T*B*512 f32 (~52 MB)

    lens_kernel<<<1, 64, 0, stream>>>(ids, lens);
    gather_kernel<<<2 * TT * 4, 256, 0, stream>>>(ids, lens, w_ih_f, b_ih_f, b_hh_f,
                                                  w_ih_b, b_ih_b, b_hh_b, gi);
    scan_kernel<<<8, 256, 65536, stream>>>(gi, w_hh_f, w_hh_b, lens, rep);
    fc_kernel<<<1, 128, 0, stream>>>(rep, fc_w, fc_b, out);
}
